// Contrast_Activate_29068338659550
// MI455X (gfx1250) — compile-verified, matrix operand order checked by reading, not run on hardware
//
#include <hip/hip_runtime.h>
#include <math.h>

typedef __attribute__((ext_vector_type(16))) _Float16 v16h;
typedef __attribute__((ext_vector_type(8)))  _Float16 v8h;
typedef __attribute__((ext_vector_type(8)))  float    v8f;
typedef __attribute__((ext_vector_type(4)))  unsigned int u32x4;
typedef __attribute__((ext_vector_type(8)))  unsigned int u32x8;

#define HH 8
#define BB 16
#define NN 1024
#define DD 256
#define DK 32
#define SLAB_ROWS 512   // half of the 1024-key slab staged in LDS via TDM

// ---- WMMA fragment builders (CDNA5 16-bit A 16x32 layout, wave32) ----
// lane 0-15: rows 0-15, K-half 0 ; lane 16-31: rows 0-15, K-half 1
// VGPR v<4 : K = 2v,2v+1 (+8*half) ; v>=4 : K = 16+2(v-4) (+8*half)
// => per lane, the 16 halfs are two contiguous 16B runs: rp[0..7], rp[16..23]
__device__ inline v16h frag_from_f32(const float* base, int ld, int koff) {
  int lane = threadIdx.x & 31;
  int half = lane >> 4, row = lane & 15;
  const float* rp = base + (size_t)row * ld + koff + half * 8;
  v16h a;
#pragma unroll
  for (int v = 0; v < 8; ++v) {
    a[v]     = (_Float16)rp[v];
    a[8 + v] = (_Float16)rp[16 + v];
  }
  return a;
}

__device__ inline v16h frag_from_f16(const _Float16* base, int ld) {
  int lane = threadIdx.x & 31;
  int half = lane >> 4, row = lane & 15;
  const _Float16* rp = base + (size_t)row * ld + half * 8;
  v8h c0 = *(const v8h*)(rp);        // b128
  v8h c1 = *(const v8h*)(rp + 16);   // b128
  v16h a;
#pragma unroll
  for (int v = 0; v < 8; ++v) { a[v] = c0[v]; a[8 + v] = c1[v]; }
  return a;
}

// ---- Kernel 1: projection GEMM  Y = X @ W^T + bias, output f16 (b,H,n,dk) ----
__global__ __launch_bounds__(32)
void proj_wmma(const float* __restrict__ X, const float* __restrict__ W,
               const float* __restrict__ bias, _Float16* __restrict__ Yh) {
  int m0 = blockIdx.x * 16;
  int o0 = blockIdx.y * 16;
  v8f acc = {};
  for (int kt = 0; kt < DD; kt += 32) {
    v16h a = frag_from_f32(X + (size_t)m0 * DD, DD, kt);  // A: 16 rows of X
    v16h b = frag_from_f32(W + (size_t)o0 * DD, DD, kt);  // B(K=c,N=o)=W[o,c]
    acc = __builtin_amdgcn_wmma_f32_16x16x32_f16(false, a, false, b,
                                                 (short)0, acc, false, false);
  }
  int lane = threadIdx.x & 31;
  int half = lane >> 4, nI = lane & 15;
  int go = o0 + nI;
  float bv = bias[go];
  int h = go >> 5, c = go & 31;
#pragma unroll
  for (int r = 0; r < 8; ++r) {
    int gm = m0 + r + 8 * half;             // global row in [0,16384)
    int b_ = gm >> 10, nr = gm & 1023;
    Yh[(((size_t)b_ * HH + h) * NN + nr) * DK + c] = (_Float16)(acc[r] + bv);
  }
}

// ---- Kernel 2: key_global[b,h,c] = mean_n k_l[b,h,n,c] * value[b,n] ----
__global__ __launch_bounds__(32)
void kg_kernel(const _Float16* __restrict__ k_h, const float* __restrict__ value,
               float* __restrict__ kg) {
  int bh = blockIdx.x;
  int b = bh >> 3;
  int c = threadIdx.x;
  const _Float16* kp = k_h + (size_t)bh * NN * DK + c;
  const float* vp = value + (size_t)b * NN;
  float s = 0.f;
  for (int i = 0; i < NN; ++i) s += (float)kp[(size_t)i * DK] * vp[i];
  kg[bh * DK + c] = s * (1.0f / NN);
}

// ---- TDM: stage a 512x32 f16 tile (32KB) from global into LDS ----
// D# per CDNA5 ISA: group0 {count=1, lds_addr,
// global_addr, type=2}; group1 {data_size=1(2B), tensor 32x512 stride 32,
// tile 32x512}.
__device__ inline void tdm_load_slab(const _Float16* gsrc, unsigned lds_base) {
  unsigned long long ga = (unsigned long long)gsrc;
  u32x4 g0;
  g0[0] = 1u;                                   // count=1, no gather
  g0[1] = lds_base;                             // lds_addr
  g0[2] = (unsigned)ga;                         // global_addr[31:0]
  g0[3] = (unsigned)((ga >> 32) & 0x01FFFFFFu)  // global_addr[56:32]
        | 0x80000000u;                          // type=2 ("image")
  u32x8 g1;
  g1[0] = 0x00010000u;        // workgroup_mask=0, data_size=1 (2 bytes)
  g1[1] = 32u << 16;          // tensor_dim0 = 32 (bits 79:48 low part)
  g1[2] = (unsigned)SLAB_ROWS << 16;  // tensor_dim1[15:0] at bits 95:80
  g1[3] = 32u << 16;          // tile_dim0 = 32 at bits 127:112
  g1[4] = (unsigned)SLAB_ROWS;        // tile_dim1 at bits 143:128
  g1[5] = 32u;                // tensor_dim0_stride[31:0] = 32 elements
  g1[6] = 0u;
  g1[7] = 0u;
  asm volatile("s_wait_dscnt 0x0" ::: "memory");  // prior LDS reads done
  asm volatile("tensor_load_to_lds %0, %1" :: "s"(g0), "s"(g1) : "memory");
  __builtin_amdgcn_s_wait_tensorcnt(0);
}

// ---- Kernel 3: two-pass softmax attention + "whole" blend -> out[...,0] ----
__global__ __launch_bounds__(32)
void attn_kernel(const _Float16* __restrict__ q_h, const _Float16* __restrict__ k_h,
                 const float* __restrict__ kg, const float* __restrict__ value,
                 const float* __restrict__ lamda, float* __restrict__ out) {
  __shared__ _Float16 kslab[SLAB_ROWS * DK];    // 32 KB
  int b = blockIdx.x >> 6, qt = blockIdx.x & 63;
  int q0 = qt * 16;
  int lane = threadIdx.x & 31, half = lane >> 4, nI = lane & 15;
  const float scale = 0.17677669529663689f;     // 1/sqrt(32)
  float lam = lamda[0];
  unsigned lds_base = (unsigned)(unsigned long long)(const void*)kslab;
  float fsum[8];
#pragma unroll
  for (int r = 0; r < 8; ++r) fsum[r] = 0.f;
  v8f zc = {};

  for (int h = 0; h < HH; ++h) {
    const _Float16* qb = q_h + (((size_t)b * HH + h) * NN + q0) * DK;
    v16h a = frag_from_f16(qb, DK);
    if (h + 1 < HH) __builtin_prefetch(qb + (size_t)NN * DK, 0, 1);

    // whole = relu(dot(q_row, kg) * scale); rows indexed by nI
    const float* kgp = kg + (b * HH + h) * DK;
    float wdot = 0.f;
#pragma unroll
    for (int v = 0; v < 8; ++v) {
      int kk = v + 8 * half;
      wdot += (float)a[v] * kgp[kk] + (float)a[8 + v] * kgp[16 + kk];
    }
    wdot += __shfl_xor(wdot, 16, 32);
    float whole = fmaxf(wdot * scale, 0.f);     // valid for row == nI

    const _Float16* kb0 = k_h + ((size_t)b * HH + h) * NN * DK;

    // ---- pass 1: row maxima (no in-loop cross-lane ops) ----
    float mx_r[8];
#pragma unroll
    for (int r = 0; r < 8; ++r) mx_r[r] = -1e30f;
    for (int sh = 0; sh < NN / SLAB_ROWS; ++sh) {
      tdm_load_slab(kb0 + (size_t)sh * SLAB_ROWS * DK, lds_base);
      for (int kt = 0; kt < SLAB_ROWS / 16; ++kt) {
        v16h bf = frag_from_f16(kslab + kt * 16 * DK, DK);
        v8f S = __builtin_amdgcn_wmma_f32_16x16x32_f16(false, a, false, bf,
                                                       (short)0, zc, false, false);
#pragma unroll
        for (int r = 0; r < 8; ++r) mx_r[r] = fmaxf(mx_r[r], S[r] * scale);
      }
    }
#pragma unroll
    for (int r = 0; r < 8; ++r) {
      float mx = mx_r[r];
      mx = fmaxf(mx, __shfl_xor(mx, 1, 32));
      mx = fmaxf(mx, __shfl_xor(mx, 2, 32));
      mx = fmaxf(mx, __shfl_xor(mx, 4, 32));
      mx = fmaxf(mx, __shfl_xor(mx, 8, 32));
      mx_r[r] = mx;                              // true row max, all lanes
    }

    // ---- pass 2: sum(exp) and sum(exp*value) (no in-loop cross-lane ops) ----
    float l_r[8], acc_r[8];
#pragma unroll
    for (int r = 0; r < 8; ++r) { l_r[r] = 0.f; acc_r[r] = 0.f; }
    for (int sh = 0; sh < NN / SLAB_ROWS; ++sh) {
      tdm_load_slab(kb0 + (size_t)sh * SLAB_ROWS * DK, lds_base);
      for (int kt = 0; kt < SLAB_ROWS / 16; ++kt) {
        v16h bf = frag_from_f16(kslab + kt * 16 * DK, DK);
        v8f S = __builtin_amdgcn_wmma_f32_16x16x32_f16(false, a, false, bf,
                                                       (short)0, zc, false, false);
        float vv = value[(size_t)b * NN + sh * SLAB_ROWS + kt * 16 + nI];
#pragma unroll
        for (int r = 0; r < 8; ++r) {
          float p = __expf(S[r] * scale - mx_r[r]);
          l_r[r] += p;
          acc_r[r] += p * vv;
        }
      }
    }
#pragma unroll
    for (int r = 0; r < 8; ++r) {
      float ps = l_r[r], pv = acc_r[r];
      ps += __shfl_xor(ps, 1, 32); ps += __shfl_xor(ps, 2, 32);
      ps += __shfl_xor(ps, 4, 32); ps += __shfl_xor(ps, 8, 32);
      pv += __shfl_xor(pv, 1, 32); pv += __shfl_xor(pv, 2, 32);
      pv += __shfl_xor(pv, 4, 32); pv += __shfl_xor(pv, 8, 32);
      float part = pv / ps;
      float wrow = __shfl(whole, r + 8 * half, 32);   // whole for row M=r+8*half
      fsum[r] += ((1.f - lam) * part + lam * wrow) * 0.125f;
    }
  }
  if (nI == 0) {
#pragma unroll
    for (int r = 0; r < 8; ++r) {
      int row = q0 + r + 8 * half;
      out[((size_t)b * NN + row) * 8] = fsum[r];
    }
  }
}

// ---- qT[b,c,i] = query[b,i,c] ----
__global__ void transpose_qT(const float* __restrict__ q, float* __restrict__ qT) {
  size_t idx = (size_t)blockIdx.x * 256 + threadIdx.x;
  if (idx >= (size_t)BB * DD * NN) return;
  int i = (int)(idx % NN);
  size_t t = idx / NN;
  int c = (int)(t % DD);
  int b = (int)(t / DD);
  qT[idx] = q[((size_t)b * NN + i) * DD + c];
}

// ---- broadcast mu0 (1,D,K) -> mu (B,D,K) ----
__global__ void bcast_mu(const float* __restrict__ mu0, float* __restrict__ mu, int K) {
  int idx = blockIdx.x * 256 + threadIdx.x;
  int tot = BB * DD * K;
  if (idx >= tot) return;
  mu[idx] = mu0[idx % (DD * K)];
}

// ---- z[b,i,k] = softmax_k(kappa * sum_c node[b,c,i]*mu[b,c,k]) ----
// (column normalization of z cancels under the later L2-norm of mu, so skipped)
__global__ void pmms_z_kernel(const float* __restrict__ node, const float* __restrict__ mu,
                              float* __restrict__ z, int C, int Nn, int K, float kappa) {
  int b = blockIdx.y;
  int i = blockIdx.x * 256 + threadIdx.x;
  if (i >= Nn) return;
  float d[5];
  for (int k = 0; k < K; ++k) d[k] = 0.f;
  const float* nb = node + (size_t)b * C * Nn;
  const float* mb = mu + (size_t)b * C * K;
  for (int c = 0; c < C; ++c) {
    float nv = nb[(size_t)c * Nn + i];
    for (int k = 0; k < K; ++k) d[k] += nv * mb[c * K + k];
  }
  float mx = d[0] * kappa;
  for (int k = 1; k < K; ++k) mx = fmaxf(mx, d[k] * kappa);
  float s = 0.f, e[5];
  for (int k = 0; k < K; ++k) { e[k] = __expf(d[k] * kappa - mx); s += e[k]; }
  float inv = 1.f / s;
  for (int k = 0; k < K; ++k) z[((size_t)b * Nn + i) * K + k] = e[k] * inv;
}

// ---- mu[b,c,k] = l2norm_c( sum_i node[b,c,i]*z[b,i,k] ) ----
__global__ __launch_bounds__(256)
void pmms_mu_kernel(const float* __restrict__ node, const float* __restrict__ z,
                    float* __restrict__ mu, int C, int Nn, int K) {
  int bk = blockIdx.x;
  int b = bk / K, k = bk % K;
  int c = threadIdx.x;  // C == 256
  const float* nb = node + ((size_t)b * C + c) * Nn;
  const float* zb = z + (size_t)b * Nn * K + k;
  float s = 0.f;
  for (int i = 0; i < Nn; ++i) s += nb[i] * zb[(size_t)i * K];
  __shared__ float red[256];
  red[c] = s * s;
  __syncthreads();
  for (int off = 128; off > 0; off >>= 1) {
    if (c < off) red[c] += red[c + off];
    __syncthreads();
  }
  float nrm = sqrtf(red[0]);
  mu[((size_t)b * C + c) * K + k] = s / (1e-6f + nrm);
}

// ---- node weight: conv3x3 -> BN -> relu -> conv3x3 -> sigmoid gate ----
__global__ __launch_bounds__(256)
void node_weight_kernel(const float* __restrict__ P, int CH,
                        const float* __restrict__ w1, const float* __restrict__ b1,
                        const float* __restrict__ g,  const float* __restrict__ be,
                        const float* __restrict__ w2, const float* __restrict__ b2,
                        float* __restrict__ out, int outoff) {
  int img = blockIdx.x;
  int b = img / CH, ch = img % CH;
  __shared__ float im[34 * 34];
  __shared__ float W1s[144], W2s[144], B1s[16], Gs[16], BEs[16];
  int t = threadIdx.x;
  for (int idx = t; idx < 34 * 34; idx += 256) im[idx] = 0.f;
  if (t < 144) { W1s[t] = w1[t]; W2s[t] = w2[t]; }
  if (t < 16)  { B1s[t] = b1[t]; Gs[t] = g[t]; BEs[t] = be[t]; }
  __syncthreads();
  for (int p = t; p < 1024; p += 256) {
    int y = p >> 5, x = p & 31;
    im[(y + 1) * 34 + (x + 1)] = P[((size_t)b * NN + p) * CH + ch];
  }
  __syncthreads();
  float bnscale = rsqrtf(1.0f + 1e-5f);
  float b2v = b2[0];
  for (int p = t; p < 1024; p += 256) {
    int y = p >> 5, x = p & 31;
    float s2 = b2v;
    for (int dy2 = -1; dy2 <= 1; ++dy2) {
      int yy = y + dy2; if (yy < 0 || yy >= 32) continue;
      for (int dx2 = -1; dx2 <= 1; ++dx2) {
        int xx = x + dx2; if (xx < 0 || xx >= 32) continue;
        float wsum = 0.f;
        int base = yy * 34 + xx;  // top-left of 3x3 window in padded coords
        for (int oc = 0; oc < 16; ++oc) {
          const float* wp = &W1s[oc * 9];
          float c1 = B1s[oc]
            + im[base]      * wp[0] + im[base + 1]  * wp[1] + im[base + 2]  * wp[2]
            + im[base + 34] * wp[3] + im[base + 35] * wp[4] + im[base + 36] * wp[5]
            + im[base + 68] * wp[6] + im[base + 69] * wp[7] + im[base + 70] * wp[8];
          float hbn = fmaxf(Gs[oc] * (c1 * bnscale) + BEs[oc], 0.f);
          wsum += W2s[oc * 9 + (dy2 + 1) * 3 + (dx2 + 1)] * hbn;
        }
        s2 += wsum;
      }
    }
    float sig = 1.f / (1.f + __expf(-s2));
    out[((size_t)b * NN + p) * 8 + outoff + ch] = im[(y + 1) * 34 + (x + 1)] * sig;
  }
}

extern "C" void kernel_launch(void* const* d_in, const int* in_sizes, int n_in,
                              void* d_out, int out_size, void* d_ws, size_t ws_size,
                              hipStream_t stream) {
  const float* query = (const float*)d_in[0];
  const float* key   = (const float*)d_in[1];
  const float* value = (const float*)d_in[2];
  const float* lamda = (const float*)d_in[3];
  const float* W0    = (const float*)d_in[4];
  const float* b0    = (const float*)d_in[5];
  const float* W1    = (const float*)d_in[6];
  const float* b1    = (const float*)d_in[7];
  const float* mu_a  = (const float*)d_in[8];
  const float* mu_b  = (const float*)d_in[9];
  const float* nw1_w1 = (const float*)d_in[10];
  const float* nw1_b1 = (const float*)d_in[11];
  const float* nw1_g  = (const float*)d_in[12];
  const float* nw1_be = (const float*)d_in[13];
  const float* nw1_w2 = (const float*)d_in[14];
  const float* nw1_b2 = (const float*)d_in[15];
  const float* nw2_w1 = (const float*)d_in[16];
  const float* nw2_b1 = (const float*)d_in[17];
  const float* nw2_g  = (const float*)d_in[18];
  const float* nw2_be = (const float*)d_in[19];
  const float* nw2_w2 = (const float*)d_in[20];
  const float* nw2_b2 = (const float*)d_in[21];
  float* out = (float*)d_out;

  char* ws = (char*)d_ws;
  size_t off = 0;
  auto alloc = [&](size_t bytes) -> void* {
    void* p = ws + off;
    off = (off + bytes + 255) & ~(size_t)255;
    return p;
  };
  _Float16* q_h = (_Float16*)alloc((size_t)BB * HH * NN * DK * 2);
  _Float16* k_h = (_Float16*)alloc((size_t)BB * HH * NN * DK * 2);
  float* kg  = (float*)alloc((size_t)BB * HH * DK * 4);
  float* qT  = (float*)alloc((size_t)BB * DD * NN * 4);
  float* mu1 = (float*)alloc((size_t)BB * DD * 5 * 4);
  float* mu2 = (float*)alloc((size_t)BB * DD * 2 * 4);
  float* zbuf = (float*)alloc((size_t)BB * NN * 5 * 4);
  float* P1  = (float*)alloc((size_t)BB * NN * 5 * 4);
  float* P2  = (float*)alloc((size_t)BB * NN * 2 * 4);

  // projections (WMMA f16 -> f32 accumulate)
  proj_wmma<<<dim3(BB * NN / 16, DD / 16), 32, 0, stream>>>(query, W0, b0, q_h);
  proj_wmma<<<dim3(BB * NN / 16, DD / 16), 32, 0, stream>>>(key,   W1, b1, k_h);

  // key_global
  kg_kernel<<<BB * HH, 32, 0, stream>>>(k_h, value, kg);

  // two-pass softmax attention (TDM-staged K slabs) -> out channel 0
  attn_kernel<<<BB * 64, 32, 0, stream>>>(q_h, k_h, kg, value, lamda, out);

  // PMMS stage 1 (node = query^T, K=5)
  transpose_qT<<<(BB * DD * NN + 255) / 256, 256, 0, stream>>>(query, qT);
  bcast_mu<<<(BB * DD * 5 + 255) / 256, 256, 0, stream>>>(mu_a, mu1, 5);
  for (int s = 0; s < 10; ++s) {
    pmms_z_kernel<<<dim3((NN + 255) / 256, BB), 256, 0, stream>>>(qT, mu1, zbuf, DD, NN, 5, 20.0f);
    pmms_mu_kernel<<<BB * 5, 256, 0, stream>>>(qT, zbuf, mu1, DD, NN, 5);
  }
  pmms_z_kernel<<<dim3((NN + 255) / 256, BB), 256, 0, stream>>>(qT, mu1, P1, DD, NN, 5, 1.0f);

  // PMMS stage 2 (node = mu1 (B,256,5), K=2)
  bcast_mu<<<(BB * DD * 2 + 255) / 256, 256, 0, stream>>>(mu_b, mu2, 2);
  for (int s = 0; s < 10; ++s) {
    pmms_z_kernel<<<dim3(1, BB), 256, 0, stream>>>(mu1, mu2, zbuf, DD, 5, 2, 20.0f);
    pmms_mu_kernel<<<BB * 2, 256, 0, stream>>>(mu1, zbuf, mu2, DD, 5, 2);
  }
  pmms_z_kernel<<<dim3((NN + 255) / 256, BB), 256, 0, stream>>>(qT, mu2, P2, DD, NN, 2, 1.0f);

  // node-weight gating -> out channels 1..5 and 6..7
  node_weight_kernel<<<BB * 5, 256, 0, stream>>>(P1, 5, nw1_w1, nw1_b1, nw1_g, nw1_be,
                                                 nw1_w2, nw1_b2, out, 1);
  node_weight_kernel<<<BB * 2, 256, 0, stream>>>(P2, 2, nw2_w1, nw2_b1, nw2_g, nw2_be,
                                                 nw2_w2, nw2_b2, out, 6);
  (void)in_sizes; (void)n_in; (void)out_size; (void)ws_size;
}
